// MllamaTextSelfAttention_13228499272294
// MI455X (gfx1250) — matrix operand order repaired and checked by reading, not compiled, not run on hardware
//
#include <hip/hip_runtime.h>

typedef __attribute__((ext_vector_type(16))) _Float16 v16h;
typedef __attribute__((ext_vector_type(8)))  _Float16 v8h;
typedef __attribute__((ext_vector_type(8)))  float    v8f;

#define S_LEN 2048
#define HID   4096
#define NH    32
#define NKV   8
#define HD    128

// ---------------------------------------------------------------- helpers ---
__device__ inline v16h cat8(v8h lo, v8h hi) {
  return __builtin_shufflevector(lo, hi, 0,1,2,3,4,5,6,7,8,9,10,11,12,13,14,15);
}

__device__ inline v8f zero8() {
  v8f z;
  #pragma unroll
  for (int i = 0; i < 8; ++i) z[i] = 0.f;
  return z;
}

__device__ inline v8h pack8(float4 a, float4 b) {
  v8h o;
  o[0] = (_Float16)a.x; o[1] = (_Float16)a.y; o[2] = (_Float16)a.z; o[3] = (_Float16)a.w;
  o[4] = (_Float16)b.x; o[5] = (_Float16)b.y; o[6] = (_Float16)b.z; o[7] = (_Float16)b.w;
  return o;
}

__device__ inline v8f wmma32(v16h a, v16h b, v8f c) {
  return __builtin_amdgcn_wmma_f32_16x16x32_f16(false, a, false, b, (short)0, c,
                                                false, false);
}

// A fragment 16x32: lanes 0-15 hold M=0..15 K{0-7,16-23}; lanes 16-31 K{8-15,24-31}
__device__ inline v16h frag_a(const _Float16* base, int stride, int m0, int k0) {
  const int lane = threadIdx.x & 31;
  const int hf = lane >> 4, mr = lane & 15;
  const _Float16* row = base + (size_t)(m0 + mr) * stride + k0;
  v8h lo = *(const v8h*)(row + hf * 8);
  v8h hi = *(const v8h*)(row + 16 + hf * 8);
  return cat8(lo, hi);
}

// B fragment 32x16 from tile stored rows = output-col n, contiguous along K
__device__ inline v16h frag_b(const _Float16* base, int stride, int n0, int k0) {
  const int lane = threadIdx.x & 31;
  const _Float16* row = base + (size_t)(n0 + (lane & 15)) * stride + k0 + (lane >> 4) * 8;
  v8h lo = *(const v8h*)(row);
  v8h hi = *(const v8h*)(row + 16);
  return cat8(lo, hi);
}

// CDNA5 async global->LDS copy (16B per lane), tracked on ASYNCcnt.
// lds_off is a byte offset relative to LDS base (first shared array = offset 0).
__device__ inline void async_copy16(unsigned int lds_off, const void* g) {
  asm volatile("global_load_async_to_lds_b128 %0, %1, off"
               :: "v"(lds_off), "v"(g) : "memory");
}
__device__ inline void wait_async0() {
  asm volatile("s_wait_asynccnt 0x0" ::: "memory");
}

// ------------------------------------------------------------ QKV GEMM ------
// C[s, n] = sum_m hs[s,m] * W[n,m]; workgroup tile 256x64, wave tile 32x64,
// double-buffered LDS with register prefetch of the next K-slab.
__global__ __launch_bounds__(256, 2)
void qkv_gemm(const float* __restrict__ hs, const float* __restrict__ wq,
              const float* __restrict__ wk, const float* __restrict__ wv,
              float* __restrict__ Qf, float* __restrict__ Kf,
              float* __restrict__ Vf) {
  __shared__ alignas(16) _Float16 As[2][256 * 40];
  __shared__ alignas(16) _Float16 Bs[2][64 * 40];
  const int tid = threadIdx.x;
  const int w = tid >> 5, lane = tid & 31, lh = lane >> 4, ln = lane & 15;
  const int m0 = blockIdx.y * 256;
  const int n0 = blockIdx.x * 64;

  const float* W; float* Out; int ostride, oc0;
  if (n0 < NH * HD)                 { W = wq + (size_t)n0 * HID;                    Out = Qf; ostride = NH * HD;  oc0 = n0; }
  else if (n0 < NH * HD + NKV * HD) { W = wk + (size_t)(n0 - NH * HD) * HID;        Out = Kf; ostride = NKV * HD; oc0 = n0 - NH * HD; }
  else                              { W = wv + (size_t)(n0 - NH*HD - NKV*HD) * HID; Out = Vf; ostride = NKV * HD; oc0 = n0 - NH*HD - NKV*HD; }

  float4 ra[4][2], rb[2];
  auto load_regs = [&](int kk) {       // batch all global loads (one wait later)
    #pragma unroll
    for (int it = 0; it < 4; ++it) {
      int idx = tid + it * 256, r = idx >> 2, c8 = (idx & 3) * 8;
      const float* p = hs + (size_t)(m0 + r) * HID + kk + c8;
      ra[it][0] = *(const float4*)p;
      ra[it][1] = *(const float4*)(p + 4);
    }
    {
      int r = tid >> 2, c8 = (tid & 3) * 8;
      const float* p = W + (size_t)r * HID + kk + c8;
      rb[0] = *(const float4*)p;
      rb[1] = *(const float4*)(p + 4);
    }
  };
  auto store_regs = [&](int buf) {     // fp32 -> f16, 16B LDS stores
    #pragma unroll
    for (int it = 0; it < 4; ++it) {
      int idx = tid + it * 256, r = idx >> 2, c8 = (idx & 3) * 8;
      *(v8h*)(&As[buf][0] + r * 40 + c8) = pack8(ra[it][0], ra[it][1]);
    }
    {
      int r = tid >> 2, c8 = (tid & 3) * 8;
      *(v8h*)(&Bs[buf][0] + r * 40 + c8) = pack8(rb[0], rb[1]);
    }
  };

  v8f acc[2][4];
  #pragma unroll
  for (int i = 0; i < 2; ++i)
    #pragma unroll
    for (int g = 0; g < 4; ++g) acc[i][g] = zero8();

  load_regs(0);
  int buf = 0;
  for (int kk = 0; kk < HID; kk += 32) {
    store_regs(buf);
    __syncthreads();
    if (kk + 32 < HID) load_regs(kk + 32);   // overlaps with WMMA below
    v16h a0 = frag_a(&As[buf][0], 40, w * 32, 0);
    v16h a1 = frag_a(&As[buf][0], 40, w * 32 + 16, 0);
    v16h b0 = frag_b(&Bs[buf][0], 40,  0, 0);
    v16h b1 = frag_b(&Bs[buf][0], 40, 16, 0);
    v16h b2 = frag_b(&Bs[buf][0], 40, 32, 0);
    v16h b3 = frag_b(&Bs[buf][0], 40, 48, 0);
    acc[0][0] = wmma32(a0, b0, acc[0][0]);
    acc[0][1] = wmma32(a0, b1, acc[0][1]);
    acc[0][2] = wmma32(a0, b2, acc[0][2]);
    acc[0][3] = wmma32(a0, b3, acc[0][3]);
    acc[1][0] = wmma32(a1, b0, acc[1][0]);
    acc[1][1] = wmma32(a1, b1, acc[1][1]);
    acc[1][2] = wmma32(a1, b2, acc[1][2]);
    acc[1][3] = wmma32(a1, b3, acc[1][3]);
    buf ^= 1;
  }

  #pragma unroll
  for (int i = 0; i < 2; ++i) {
    const int rbase = m0 + w * 32 + i * 16 + lh * 8;
    #pragma unroll
    for (int r = 0; r < 8; ++r) {
      float* op = Out + (size_t)(rbase + r) * ostride + oc0 + ln;
      op[0] = acc[i][0][r]; op[16] = acc[i][1][r];
      op[32] = acc[i][2][r]; op[48] = acc[i][3][r];
    }
  }
}

// ------------------------------------------------- RoPE + f16 repack --------
__global__ __launch_bounds__(256)
void rope_convert(const float* __restrict__ Qf, const float* __restrict__ Kf,
                  const float* __restrict__ Vf, const float* __restrict__ cosT,
                  const float* __restrict__ sinT, _Float16* __restrict__ Qh,
                  _Float16* __restrict__ Kh, _Float16* __restrict__ Vh) {
  const int NQ = NH * S_LEN * 64;
  const int NK = NKV * S_LEN * 64;
  const int NV = NKV * S_LEN * HD;
  int tid = blockIdx.x * 256 + threadIdx.x;
  if (tid < NQ) {
    int d = tid & 63, s = (tid >> 6) & (S_LEN - 1), h = tid >> 17;
    float x1 = Qf[(size_t)s * (NH * HD) + h * HD + d];
    float x2 = Qf[(size_t)s * (NH * HD) + h * HD + d + 64];
    float c = cosT[s * HD + d], sn = sinT[s * HD + d];
    size_t o = ((size_t)h * S_LEN + s) * HD + d;
    Qh[o]      = (_Float16)(x1 * c - x2 * sn);
    Qh[o + 64] = (_Float16)(x2 * c + x1 * sn);
  } else if (tid < NQ + NK) {
    int t = tid - NQ;
    int d = t & 63, s = (t >> 6) & (S_LEN - 1), h = t >> 17;
    float x1 = Kf[(size_t)s * (NKV * HD) + h * HD + d];
    float x2 = Kf[(size_t)s * (NKV * HD) + h * HD + d + 64];
    float c = cosT[s * HD + d], sn = sinT[s * HD + d];
    size_t o = ((size_t)h * S_LEN + s) * HD + d;
    Kh[o]      = (_Float16)(x1 * c - x2 * sn);
    Kh[o + 64] = (_Float16)(x2 * c + x1 * sn);
  } else if (tid < NQ + NK + NV) {
    int t = tid - NQ - NK;
    int d = t & 127, s = (t >> 7) & (S_LEN - 1), h = t >> 18;
    Vh[((size_t)h * S_LEN + s) * HD + d] =
        (_Float16)Vf[(size_t)s * (NKV * HD) + h * HD + d];
  }
}

// --------------------------------------------------- flash attention --------
__global__ __launch_bounds__(256, 1)
void flash_attn(const _Float16* __restrict__ Qh, const _Float16* __restrict__ Kh,
                const _Float16* __restrict__ Vh, _Float16* __restrict__ AO) {
  __shared__ alignas(16) _Float16 Ks[32 * 136];    // FIRST shared array: LDS off 0
  __shared__ alignas(16) _Float16 Vt[128 * 40];    // [d][key] transposed
  __shared__ alignas(16) _Float16 Ps[8 * 16 * 40]; // per-wave P scratch
  const int tid = threadIdx.x;
  const int w = tid >> 5, lane = tid & 31, lh = lane >> 4, ln = lane & 15;
  const int h = blockIdx.x;
  const int qb = blockIdx.y * 128;
  const int kvh = h >> 2;
  const int qw = qb + w * 16;

  v16h aQ[4];
  const _Float16* Qrow = Qh + ((size_t)h * S_LEN + qw + ln) * HD;
  #pragma unroll
  for (int g = 0; g < 4; ++g) {
    v8h lo = *(const v8h*)(Qrow + g * 32 + lh * 8);
    v8h hi = *(const v8h*)(Qrow + g * 32 + 16 + lh * 8);
    aQ[g] = cat8(lo, hi);
  }

  v8f o[8];
  #pragma unroll
  for (int g = 0; g < 8; ++g) o[g] = zero8();
  float m_r[8], l_r[8];
  #pragma unroll
  for (int r = 0; r < 8; ++r) { m_r[r] = -1e30f; l_r[r] = 0.f; }

  const int ntiles = (qb + 128) >> 5;
  const float scale = 0.088388347648318447f;

  for (int j = 0; j < ntiles; ++j) {
    const int key0 = j * 32;
    const _Float16* Kbase = Kh + ((size_t)kvh * S_LEN + key0) * HD;
    const _Float16* Vbase = Vh + ((size_t)kvh * S_LEN + key0) * HD;
    v8h vv[2];
    #pragma unroll
    for (int it = 0; it < 2; ++it) {               // K tile: async -> LDS (ASYNCcnt)
      int idx = tid + it * 256, kr = idx >> 4, dc = (idx & 15) * 8;
      async_copy16((unsigned)((kr * 136 + dc) * 2), Kbase + (size_t)kr * HD + dc);
      vv[it] = *(const v8h*)(Vbase + (size_t)kr * HD + dc);
    }
    #pragma unroll
    for (int it = 0; it < 2; ++it) {               // V tile: transpose into LDS
      int idx = tid + it * 256, kr = idx >> 4, dc = (idx & 15) * 8;
      #pragma unroll
      for (int i = 0; i < 8; ++i) Vt[(dc + i) * 40 + kr] = vv[it][i];
    }
    wait_async0();
    __syncthreads();

    if (key0 <= qw + 15) {                         // wave-uniform causal skip
      v8f s0 = zero8(), s1 = zero8();
      v16h bt[4];
      #pragma unroll
      for (int g = 0; g < 4; ++g) bt[g] = frag_b(Ks, 136, 0, g * 32);
      #pragma unroll
      for (int g = 0; g < 4; ++g) s0 = wmma32(aQ[g], bt[g], s0);
      #pragma unroll
      for (int g = 0; g < 4; ++g) bt[g] = frag_b(Ks, 136, 16, g * 32);
      #pragma unroll
      for (int g = 0; g < 4; ++g) s1 = wmma32(aQ[g], bt[g], s1);

      _Float16* P = Ps + w * (16 * 40);
      #pragma unroll
      for (int r = 0; r < 8; ++r) {                // online softmax per q-row
        const int q = qw + r + lh * 8;
        const bool u0 = (key0 + ln <= q), u1 = (key0 + 16 + ln <= q);
        float v0 = u0 ? s0[r] * scale : -1e30f;
        float v1 = u1 ? s1[r] * scale : -1e30f;
        float rm = fmaxf(v0, v1);
        #pragma unroll
        for (int mm = 1; mm < 16; mm <<= 1) rm = fmaxf(rm, __shfl_xor(rm, mm, 32));
        const float mnew  = fmaxf(m_r[r], rm);
        const float alpha = __expf(m_r[r] - mnew);
        const float e0 = u0 ? __expf(v0 - mnew) : 0.f;
        const float e1 = u1 ? __expf(v1 - mnew) : 0.f;
        float rs = e0 + e1;
        #pragma unroll
        for (int mm = 1; mm < 16; mm <<= 1) rs += __shfl_xor(rs, mm, 32);
        l_r[r] = l_r[r] * alpha + rs;
        m_r[r] = mnew;
        #pragma unroll
        for (int g = 0; g < 8; ++g) o[g][r] *= alpha;
        P[(r + lh * 8) * 40 + ln]      = (_Float16)e0;
        P[(r + lh * 8) * 40 + 16 + ln] = (_Float16)e1;
      }
      __asm__ volatile("s_wait_dscnt 0" ::: "memory");
      v16h aP = frag_a(P, 40, 0, 0);
      v16h bv[4];
      #pragma unroll
      for (int g = 0; g < 4; ++g) bv[g] = frag_b(Vt, 40, g * 16, 0);
      #pragma unroll
      for (int g = 0; g < 4; ++g) o[g] = wmma32(aP, bv[g], o[g]);
      #pragma unroll
      for (int g = 0; g < 4; ++g) bv[g] = frag_b(Vt, 40, (4 + g) * 16, 0);
      #pragma unroll
      for (int g = 0; g < 4; ++g) o[4 + g] = wmma32(aP, bv[g], o[4 + g]);
    }
    __syncthreads();
  }

  #pragma unroll
  for (int r = 0; r < 8; ++r) {
    const float inv = 1.0f / l_r[r];
    const size_t row = (size_t)(qw + r + lh * 8) * (NH * HD) + (size_t)h * HD + ln;
    #pragma unroll
    for (int g = 0; g < 8; ++g)
      AO[row + g * 16] = (_Float16)(o[g][r] * inv);
  }
}

// ------------------------------------------------------- output proj --------
// out[s, m] = sum_n AO[s,n] * wo[m,n]; A-tile staged via async LDS copies.
__global__ __launch_bounds__(256, 2)
void oproj_gemm(const _Float16* __restrict__ AO, const float* __restrict__ wo,
                float* __restrict__ out) {
  __shared__ alignas(16) _Float16 As[2][256 * 40]; // FIRST shared array: off 0
  __shared__ alignas(16) _Float16 Bs[2][64 * 40];
  const int tid = threadIdx.x;
  const int w = tid >> 5, lane = tid & 31, lh = lane >> 4, ln = lane & 15;
  const int m0 = blockIdx.y * 256;
  const int n0 = blockIdx.x * 64;

  auto async_A = [&](int kk, int buf) {            // f16 copy: async -> LDS
    #pragma unroll
    for (int it = 0; it < 4; ++it) {
      int idx = tid + it * 256, r = idx >> 2, c8 = (idx & 3) * 8;
      async_copy16((unsigned)(buf * (256 * 40 * 2) + (r * 40 + c8) * 2),
                   AO + (size_t)(m0 + r) * (NH * HD) + kk + c8);
    }
  };
  float4 rb[2];
  auto load_B = [&](int kk) {
    int r = tid >> 2, c8 = (tid & 3) * 8;
    const float* p = wo + (size_t)(n0 + r) * (NH * HD) + kk + c8;
    rb[0] = *(const float4*)p;
    rb[1] = *(const float4*)(p + 4);
  };
  auto store_B = [&](int buf) {
    int r = tid >> 2, c8 = (tid & 3) * 8;
    *(v8h*)(&Bs[buf][0] + r * 40 + c8) = pack8(rb[0], rb[1]);
  };

  v8f acc[2][4];
  #pragma unroll
  for (int i = 0; i < 2; ++i)
    #pragma unroll
    for (int g = 0; g < 4; ++g) acc[i][g] = zero8();

  async_A(0, 0);
  load_B(0);
  int buf = 0;
  for (int kk = 0; kk < NH * HD; kk += 32) {
    store_B(buf);
    wait_async0();                                  // A[buf] resident in LDS
    __syncthreads();
    if (kk + 32 < NH * HD) { async_A(kk + 32, buf ^ 1); load_B(kk + 32); }
    v16h a0 = frag_a(&As[buf][0], 40, w * 32, 0);
    v16h a1 = frag_a(&As[buf][0], 40, w * 32 + 16, 0);
    v16h b0 = frag_b(&Bs[buf][0], 40,  0, 0);
    v16h b1 = frag_b(&Bs[buf][0], 40, 16, 0);
    v16h b2 = frag_b(&Bs[buf][0], 40, 32, 0);
    v16h b3 = frag_b(&Bs[buf][0], 40, 48, 0);
    acc[0][0] = wmma32(a0, b0, acc[0][0]);
    acc[0][1] = wmma32(a0, b1, acc[0][1]);
    acc[0][2] = wmma32(a0, b2, acc[0][2]);
    acc[0][3] = wmma32(a0, b3, acc[0][3]);
    acc[1][0] = wmma32(a1, b0, acc[1][0]);
    acc[1][1] = wmma32(a1, b1, acc[1][1]);
    acc[1][2] = wmma32(a1, b2, acc[1][2]);
    acc[1][3] = wmma32(a1, b3, acc[1][3]);
    buf ^= 1;
  }

  #pragma unroll
  for (int i = 0; i < 2; ++i) {
    const int rbase = m0 + w * 32 + i * 16 + lh * 8;
    #pragma unroll
    for (int r = 0; r < 8; ++r) {
      float* op = out + (size_t)(rbase + r) * HID + n0 + ln;
      op[0] = acc[i][0][r]; op[16] = acc[i][1][r];
      op[32] = acc[i][2][r]; op[48] = acc[i][3][r];
    }
  }
}

// ---------------------------------------------------------------- launch ----
extern "C" void kernel_launch(void* const* d_in, const int* in_sizes, int n_in,
                              void* d_out, int out_size, void* d_ws, size_t ws_size,
                              hipStream_t stream) {
  (void)in_sizes; (void)n_in; (void)out_size; (void)ws_size;
  const float* hs   = (const float*)d_in[0];
  const float* cosT = (const float*)d_in[2];
  const float* sinT = (const float*)d_in[3];
  const float* wq   = (const float*)d_in[4];
  const float* wk   = (const float*)d_in[5];
  const float* wv   = (const float*)d_in[6];
  const float* wo   = (const float*)d_in[7];
  float* out = (float*)d_out;

  char* ws = (char*)d_ws;
  float* Qf = (float*)ws;       ws += (size_t)S_LEN * (NH * HD) * 4;
  float* Kf = (float*)ws;       ws += (size_t)S_LEN * (NKV * HD) * 4;
  float* Vf = (float*)ws;       ws += (size_t)S_LEN * (NKV * HD) * 4;
  _Float16* Qh = (_Float16*)ws; ws += (size_t)S_LEN * (NH * HD) * 2;
  _Float16* Kh = (_Float16*)ws; ws += (size_t)S_LEN * (NKV * HD) * 2;
  _Float16* Vh = (_Float16*)ws; ws += (size_t)S_LEN * (NKV * HD) * 2;
  _Float16* AO = (_Float16*)ws;

  qkv_gemm<<<dim3(96, 8), 256, 0, stream>>>(hs, wq, wk, wv, Qf, Kf, Vf);

  const int rope_total = NH * S_LEN * 64 + NKV * S_LEN * 64 + NKV * S_LEN * HD;
  rope_convert<<<dim3((rope_total + 255) / 256), 256, 0, stream>>>(
      Qf, Kf, Vf, cosT, sinT, Qh, Kh, Vh);

  flash_attn<<<dim3(NH, S_LEN / 128), 256, 0, stream>>>(Qh, Kh, Vh, AO);

  oproj_gemm<<<dim3(HID / 64, S_LEN / 128 * 2), 256, 0, stream>>>(AO, wo, out);
}
